// RelativeAttentionHead_19774029430824
// MI455X (gfx1250) — matrix operand order repaired and checked by reading, not compiled, not run on hardware
//
#include <hip/hip_runtime.h>

typedef __attribute__((ext_vector_type(16))) __bf16 v16bf;
typedef __attribute__((ext_vector_type(8)))  float  v8f;

namespace {
constexpr int Bn = 8;
constexpr int Tn = 2048;
constexpr int Cn = 512;
constexpr int Hn = 64;
}

union BF16x16 {
  v16bf v;
  unsigned int w[8];
  unsigned short h[16];
};

__device__ __forceinline__ unsigned short f2bf(float f) {
  unsigned int u = __float_as_uint(f);
  u += 0x7FFFu + ((u >> 16) & 1u);          // round-to-nearest-even
  return (unsigned short)(u >> 16);
}
__device__ __forceinline__ unsigned int pk2(float lo, float hi) {
  return (unsigned int)f2bf(lo) | ((unsigned int)f2bf(hi) << 16);
}
__device__ __forceinline__ void set_w(BF16x16& o, uint4 a, uint4 b) {
  o.w[0] = a.x; o.w[1] = a.y; o.w[2] = a.z; o.w[3] = a.w;
  o.w[4] = b.x; o.w[5] = b.y; o.w[6] = b.z; o.w[7] = b.w;
}
__device__ __forceinline__ v8f wmma_bf16(const BF16x16& a, const BF16x16& b, v8f c) {
  return __builtin_amdgcn_wmma_f32_16x16x32_bf16(false, a.v, false, b.v,
                                                 (short)0, c, false, false);
}

// ---------------------------------------------------------------- E -> bf16
__global__ __launch_bounds__(256) void cvt_bf16_kernel(const float* __restrict__ src,
                                                       unsigned short* __restrict__ dst,
                                                       int n) {
  int i = blockIdx.x * 256 + threadIdx.x;
  if (i < n) dst[i] = f2bf(src[i]);
}

// ------------------------------------------------- q,k,v projections (WMMA)
// block = 256 threads = 8 waves; each wave computes 16 rows of q/k/v.
// q,k stored (B*T, H) bf16 row-major; v stored transposed (B, H, T) bf16.
__global__ __launch_bounds__(256) void proj_kernel(const float* __restrict__ x,
                                                   const float* __restrict__ Wq,
                                                   const float* __restrict__ Wk,
                                                   const float* __restrict__ Wv,
                                                   unsigned short* __restrict__ qb,
                                                   unsigned short* __restrict__ kb,
                                                   unsigned short* __restrict__ vTb) {
  __shared__ __align__(16) unsigned short ldsW[3 * 64 * 32];  // [w][n][kk], 12 KB

  const int tid  = threadIdx.x;
  const int wave = tid >> 5;
  const int lane = tid & 31;
  const int m    = lane & 15;
  const int hi   = lane >> 4;
  const int rowBase = blockIdx.x * 128 + wave * 16;  // row in [0, B*T)

  v8f acc[3][4];
#pragma unroll
  for (int w = 0; w < 3; ++w)
#pragma unroll
    for (int nt = 0; nt < 4; ++nt)
      acc[w][nt] = (v8f){0.f, 0.f, 0.f, 0.f, 0.f, 0.f, 0.f, 0.f};

  for (int c0 = 0; c0 < Cn; c0 += 32) {
    __syncthreads();
    // stage W chunk (32 x 64 per matrix), transposed: ldsW[(w*64+n)*32+kk]
    for (int i = tid; i < 3 * 64 * 32; i += 256) {
      int w   = i >> 11;
      int rem = i & 2047;
      int n   = rem >> 5;
      int kk  = rem & 31;
      const float* Wp = (w == 0) ? Wq : ((w == 1) ? Wk : Wv);
      ldsW[i] = f2bf(Wp[(size_t)(c0 + kk) * Hn + n]);
    }
    __syncthreads();

    // A operand: 16 rows of x, K = c0..c0+31 (bf16 on the fly)
    BF16x16 a;
    const float* xr = x + (size_t)(rowBase + m) * Cn + c0;
    float4 f0 = *(const float4*)(xr + hi * 8);
    float4 f1 = *(const float4*)(xr + hi * 8 + 4);
    float4 f2 = *(const float4*)(xr + 16 + hi * 8);
    float4 f3 = *(const float4*)(xr + 16 + hi * 8 + 4);
    a.w[0] = pk2(f0.x, f0.y); a.w[1] = pk2(f0.z, f0.w);
    a.w[2] = pk2(f1.x, f1.y); a.w[3] = pk2(f1.z, f1.w);
    a.w[4] = pk2(f2.x, f2.y); a.w[5] = pk2(f2.z, f2.w);
    a.w[6] = pk2(f3.x, f3.y); a.w[7] = pk2(f3.z, f3.w);

#pragma unroll
    for (int w = 0; w < 3; ++w) {
#pragma unroll
      for (int nt = 0; nt < 4; ++nt) {
        const unsigned short* bp = &ldsW[((w << 6) + (nt << 4) + m) << 5];
        BF16x16 bop;
        set_w(bop, *(const uint4*)(bp + hi * 8), *(const uint4*)(bp + 16 + hi * 8));
        acc[w][nt] = wmma_bf16(a, bop, acc[w][nt]);
      }
    }
  }

  // store: C/D layout lane holds N = m, rows M = j + hi*8
#pragma unroll
  for (int nt = 0; nt < 4; ++nt) {
#pragma unroll
    for (int j = 0; j < 8; ++j) {
      int M = j + hi * 8;
      size_t row = (size_t)(rowBase + M);
      int col = nt * 16 + m;
      qb[row * Hn + col] = f2bf(acc[0][nt][j]);
      kb[row * Hn + col] = f2bf(acc[1][nt][j]);
      int b = (int)(row >> 11);      // /T
      int t = (int)(row & (Tn - 1));
      vTb[((size_t)b * Hn + col) * Tn + t] = f2bf(acc[2][nt][j]);
    }
  }
}

// ------------------------------------------- fused flash attention + rel bias
// one wave per 16-query tile; 8 waves per block; 32-key steps.
__global__ __launch_bounds__(256) void attn_kernel(const unsigned short* __restrict__ qb,
                                                   const unsigned short* __restrict__ kb,
                                                   const unsigned short* __restrict__ vTb,
                                                   const unsigned short* __restrict__ Eb,
                                                   float* __restrict__ out) {
  __shared__ __align__(16) float          ldsG[8][16 * 48];  // rel window tile
  __shared__ __align__(16) unsigned short ldsP[8][16 * 32];  // probabilities (bf16)

  const int tid  = threadIdx.x;
  const int wave = tid >> 5;
  const int lane = tid & 31;
  const int n    = lane & 15;
  const int hi   = lane >> 4;

  const int tile = blockIdx.x * 8 + wave;
  const int b    = tile >> 7;          // 128 q-tiles per batch
  const int q0   = (tile & 127) << 4;

  float* Gbuf = ldsG[wave];
  unsigned short* Pbuf = ldsP[wave];

  // Q A-operands (16 x 64, two K=32 chunks), resident for whole kernel
  BF16x16 aQ[2];
  {
    const unsigned short* qr = qb + ((size_t)(b * Tn + q0 + n)) * Hn;
#pragma unroll
    for (int s = 0; s < 2; ++s)
      set_w(aQ[s], *(const uint4*)(qr + 32 * s + hi * 16),
                   *(const uint4*)(qr + 32 * s + hi * 16 + 8));
  }

  float mrow[8], lrow[8];
  v8f acc[4];
#pragma unroll
  for (int j = 0; j < 8; ++j) { mrow[j] = -3e38f; lrow[j] = 0.f; }
#pragma unroll
  for (int nt = 0; nt < 4; ++nt)
    acc[nt] = (v8f){0.f, 0.f, 0.f, 0.f, 0.f, 0.f, 0.f, 0.f};

  const int nsteps = (q0 + 47) >> 5;
  for (int step = 0; step < nsteps; ++step) {
    const int k0 = step << 5;

    if (step + 1 < nsteps) {   // prefetch next K/V tiles
      __builtin_prefetch(kb + ((size_t)(b * Tn + k0 + 32 + n)) * Hn, 0, 1);
      __builtin_prefetch(vTb + ((size_t)(b * Hn + n)) * Tn + k0 + 32, 0, 1);
    }

    // ---- S = Q K^T  (16 x 32, two 16x16 tiles) ----
    v8f s[2];
#pragma unroll
    for (int t = 0; t < 2; ++t) {
      s[t] = (v8f){0.f, 0.f, 0.f, 0.f, 0.f, 0.f, 0.f, 0.f};
      const unsigned short* kr = kb + ((size_t)(b * Tn + k0 + t * 16 + n)) * Hn;
#pragma unroll
      for (int c = 0; c < 2; ++c) {
        BF16x16 bop;
        set_w(bop, *(const uint4*)(kr + 32 * c + hi * 16),
                   *(const uint4*)(kr + 32 * c + hi * 16 + 8));
        s[t] = wmma_bf16(aQ[c], bop, s[t]);
      }
    }

    // ---- G = Q * E_window^T (16 x 48) -> LDS ----
    const int r0 = Tn - 17 - q0 + k0;
#pragma unroll
    for (int gt = 0; gt < 3; ++gt) {
      v8f g = (v8f){0.f, 0.f, 0.f, 0.f, 0.f, 0.f, 0.f, 0.f};
      int c = gt * 16 + n;
      int r = r0 + c;
      r = (r < 0) ? 0 : ((r > Tn - 1) ? Tn - 1 : r);   // clamped rows are masked later
      const unsigned short* er = Eb + (size_t)r * Hn;
#pragma unroll
      for (int cc = 0; cc < 2; ++cc) {
        BF16x16 bop;
        set_w(bop, *(const uint4*)(er + 32 * cc + hi * 16),
                   *(const uint4*)(er + 32 * cc + hi * 16 + 8));
        g = wmma_bf16(aQ[cc], bop, g);
      }
#pragma unroll
      for (int j = 0; j < 8; ++j) Gbuf[(j + hi * 8) * 48 + c] = g[j];
    }

    // ---- gather skewed rel bias, scale, causal mask ----
    float sv[2][8];
#pragma unroll
    for (int t = 0; t < 2; ++t) {
#pragma unroll
      for (int j = 0; j < 8; ++j) {
        int M  = j + hi * 8;
        int Ng = t * 16 + n;
        float srel = Gbuf[M * 48 + (Ng - M + 16)];
        float val  = (s[t][j] + srel) * 0.125f;   // 1/sqrt(64)
        sv[t][j] = (k0 + Ng <= q0 + M) ? val : -3e38f;
      }
    }

    // ---- online softmax update ----
#pragma unroll
    for (int j = 0; j < 8; ++j) {
      float rm = fmaxf(sv[0][j], sv[1][j]);
#pragma unroll
      for (int msk = 1; msk < 16; msk <<= 1)
        rm = fmaxf(rm, __shfl_xor(rm, msk, 32));
      float mnew  = fmaxf(mrow[j], rm);
      float scale = __expf(mrow[j] - mnew);
      float p0 = __expf(sv[0][j] - mnew);
      float p1 = __expf(sv[1][j] - mnew);
      float rs = p0 + p1;
#pragma unroll
      for (int msk = 1; msk < 16; msk <<= 1)
        rs += __shfl_xor(rs, msk, 32);
      lrow[j] = lrow[j] * scale + rs;
      mrow[j] = mnew;
#pragma unroll
      for (int nt = 0; nt < 4; ++nt) acc[nt][j] *= scale;
      int M = j + hi * 8;
      Pbuf[M * 32 + n]      = f2bf(p0);
      Pbuf[M * 32 + 16 + n] = f2bf(p1);
    }

    // ---- P through LDS into A-operand layout ----
    BF16x16 pA;
    {
      const unsigned short* pr = Pbuf + n * 32;
      set_w(pA, *(const uint4*)(pr + hi * 16), *(const uint4*)(pr + hi * 16 + 8));
    }

    // ---- O += P V  (V transposed: contiguous b128 B-operand loads) ----
#pragma unroll
    for (int nt = 0; nt < 4; ++nt) {
      const unsigned short* vr = vTb + ((size_t)(b * Hn + nt * 16 + n)) * Tn + k0;
      BF16x16 bop;
      set_w(bop, *(const uint4*)(vr + hi * 16), *(const uint4*)(vr + hi * 16 + 8));
      acc[nt] = wmma_bf16(pA, bop, acc[nt]);
    }
  }

  // ---- normalize and store (B,T,H) f32 ----
#pragma unroll
  for (int nt = 0; nt < 4; ++nt) {
#pragma unroll
    for (int j = 0; j < 8; ++j) {
      int M = j + hi * 8;
      out[((size_t)(b * Tn + q0 + M)) * Hn + nt * 16 + n] = acc[nt][j] / lrow[j];
    }
  }
}

// ---------------------------------------------------------------------------
extern "C" void kernel_launch(void* const* d_in, const int* in_sizes, int n_in,
                              void* d_out, int out_size, void* d_ws, size_t ws_size,
                              hipStream_t stream) {
  const float* x  = (const float*)d_in[0];
  const float* Wq = (const float*)d_in[1];
  const float* Wk = (const float*)d_in[2];
  const float* Wv = (const float*)d_in[3];
  const float* E  = (const float*)d_in[4];
  float* out = (float*)d_out;

  unsigned short* qb  = (unsigned short*)d_ws;
  unsigned short* kb  = qb  + (size_t)Bn * Tn * Hn;
  unsigned short* vTb = kb  + (size_t)Bn * Tn * Hn;
  unsigned short* Eb  = vTb + (size_t)Bn * Tn * Hn;

  const int nE = Tn * Hn;
  cvt_bf16_kernel<<<(nE + 255) / 256, 256, 0, stream>>>(E, Eb, nE);
  proj_kernel<<<Bn * Tn / 128, 256, 0, stream>>>(x, Wq, Wk, Wv, qb, kb, vTb);
  attn_kernel<<<Bn * Tn / 16 / 8, 256, 0, stream>>>(qb, kb, vTb, Eb, out);
}
